// MolyEncoderAggP2_11416023073420
// MI455X (gfx1250) — matrix operand order repaired and checked by reading, not compiled, not run on hardware
//
#include <hip/hip_runtime.h>

// ---------------- problem constants ----------------
#define BB     32
#define MM     512
#define NN     512
#define EE     768
#define FIVE_E  (5 * EE)    // 3840
#define THREE_E (3 * EE)    // 2304
#define FIFT_E  (15 * EE)   // 11520
#define INV_SCALE 0.036084391824351615f  // 1/sqrt(768)
#define NEGBIG   (-1.0e12f)

// LDS B-tile: 64 columns x 32 K-elements, column stride padded to 40 ushorts
// (80B: 16B-aligned for b128 ds loads, bank-conflict-friendly).
#define BSTRIDE 40
#define BTILE   (64 * BSTRIDE)

typedef __attribute__((ext_vector_type(16))) __bf16 v16bf;
typedef __attribute__((ext_vector_type(8)))  float  v8f;
typedef unsigned short u16;

// ---------------- bf16 helpers ----------------
__device__ __forceinline__ u16 f2bf(float f) {
  unsigned int u = __float_as_uint(f);
  unsigned int r = (u + 0x7FFFu + ((u >> 16) & 1u)) >> 16;
  return (u16)r;
}
__device__ __forceinline__ float bf2f(u16 s) {
  return __uint_as_float(((unsigned int)s) << 16);
}

union AF {
  v16bf v;
  u16   s[16];
  uint4 q[2];
};

// A fragment (16x32 bf16, row-major source, pitch in elements).
// lane = r + 16*h ; elems 0..7 -> K = k0 + 8h + j ; elems 8..15 -> K = k0 + 16 + 8h + j
__device__ __forceinline__ v16bf frag_a_row(const u16* base, int pitch, int k0, int lane) {
  int r = lane & 15, h = lane >> 4;
  const u16* p = base + (size_t)r * pitch + k0 + h * 8;
  AF f;
  f.q[0] = *(const uint4*)(p);
  f.q[1] = *(const uint4*)(p + 16);
  return f.v;
}

// B fragment read from the staged LDS tile. elems 0..7 -> K = 8h + j ; elems 8..15 -> K = 16 + 8h + j for column c.
__device__ __forceinline__ v16bf frag_b_lds(const u16* buf, int cb, int lane) {
  int c = lane & 15, h = lane >> 4;
  const u16* p = buf + (size_t)(cb * 16 + c) * BSTRIDE + h * 8;
  AF f;
  f.q[0] = *(const uint4*)(p);
  f.q[1] = *(const uint4*)(p + 16);
  return f.v;
}

// Cooperative async stage of one 64-col x 32-K B tile into LDS.
// 256 threads, one global_load_async_to_lds_b128 each (4KB total).
// colbase points at column 0 of the 64-column block; pitch in elements.
__device__ __forceinline__ void stage_b(const u16* colbase, int pitch, int k0,
                                        u16* buf, int tid) {
  int c  = tid >> 2;           // 0..63
  int kq = (tid & 3) * 8;      // 0,8,16,24
  const u16* src = colbase + (size_t)c * pitch + k0 + kq;
  unsigned dst = (unsigned)(size_t)(buf + c * BSTRIDE + kq);
  asm volatile("global_load_async_to_lds_b128 %0, %1, off"
               :: "v"(dst), "v"(src) : "memory");
}
__device__ __forceinline__ void wait_async_le1() {
  asm volatile("s_wait_asynccnt 0x1" ::: "memory");
}
__device__ __forceinline__ void wait_async_0() {
  asm volatile("s_wait_asynccnt 0x0" ::: "memory");
}

// A fragment built on the fly from submult([hat, hs]) expansion, K dim = 5*E.
// Each 32-wide K tile lies inside one segment (768 % 32 == 0).
__device__ __forceinline__ v16bf frag_a_submult(const u16* hat_row, const u16* hs_row,
                                                int k0, int lane) {
  int r = lane & 15, h = lane >> 4;
  int seg = k0 / EE;
  int u   = (k0 % EE) + h * 8;
  const u16* ph = hat_row + (size_t)r * EE + u;
  const u16* ps = hs_row  + (size_t)r * EE + u;
  AF fa, fb, out;
  fa.q[0] = *(const uint4*)(ph);      fa.q[1] = *(const uint4*)(ph + 16);
  fb.q[0] = *(const uint4*)(ps);      fb.q[1] = *(const uint4*)(ps + 16);
#pragma unroll
  for (int j = 0; j < 16; ++j) {
    float a = bf2f(fa.s[j]), b = bf2f(fb.s[j]);
    float v;
    if      (seg == 0) v = a;
    else if (seg == 1) v = b;
    else if (seg == 2) v = a - b;
    else if (seg == 3) v = a * b;
    else { float d = a - b; v = d * d; }
    out.s[j] = f2bf(v);
  }
  return out.v;
}

// ---------------- stage 0: f32 -> bf16 (+ transposed copies) ----------------
__global__ __launch_bounds__(256) void convert_kernel(
    const float* __restrict__ c_hs, const float* __restrict__ r_hs,
    u16* __restrict__ c_bf, u16* __restrict__ r_bf,
    u16* __restrict__ cT,   u16* __restrict__ rT) {
  size_t idx = (size_t)blockIdx.x * 256 + threadIdx.x;
  const size_t total = (size_t)BB * MM * EE;
  if (idx >= total) return;
  const float* src = blockIdx.y ? r_hs : c_hs;
  u16* d  = blockIdx.y ? r_bf : c_bf;
  u16* dT = blockIdx.y ? rT   : cT;
  u16 bf = f2bf(src[idx]);
  d[idx] = bf;
  size_t b = idx / ((size_t)MM * EE);
  size_t rem = idx - b * (size_t)MM * EE;
  size_t m = rem / EE, e = rem - m * EE;
  dT[(b * EE + e) * MM + m] = bf;
}

__global__ __launch_bounds__(256) void wpret_kernel(
    const float* __restrict__ Wpre, u16* __restrict__ WT) {
  int idx = blockIdx.x * 256 + threadIdx.x;   // over E*5E
  int e = idx / FIVE_E, t = idx - e * FIVE_E;
  WT[idx] = f2bf(Wpre[(size_t)t * EE + e]);
}

// ---------------- stage 1: att = (C . R^T)/scale with pad mask ----------------
__global__ __launch_bounds__(256) void att_kernel(
    const u16* __restrict__ c_bf, const u16* __restrict__ r_bf,
    const int* __restrict__ c_ids, const int* __restrict__ r_ids,
    float* __restrict__ att) {
  __shared__ alignas(16) u16 bstage[2][BTILE];
  int b = blockIdx.z;
  int tid = threadIdx.x;
  int lane = tid & 31, w = tid >> 5;
  int m0 = blockIdx.x * 128 + w * 16;
  int n0 = blockIdx.y * 64;
  const u16* A    = c_bf + (size_t)b * MM * EE + (size_t)m0 * EE;
  const u16* Bcol = r_bf + (size_t)b * NN * EE + (size_t)n0 * EE;
  v8f z = {0.f,0.f,0.f,0.f,0.f,0.f,0.f,0.f};
  v8f acc[4] = {z, z, z, z};
  const int KIT = EE / 32;  // 24
  stage_b(Bcol, EE, 0, bstage[0], tid);
  for (int ki = 0; ki < KIT; ++ki) {
    int k0 = ki * 32;
    if (ki + 1 < KIT) { stage_b(Bcol, EE, k0 + 32, bstage[(ki + 1) & 1], tid); wait_async_le1(); }
    else              { wait_async_0(); }
    __syncthreads();
    v16bf a = frag_a_row(A, EE, k0, lane);
    const u16* buf = bstage[ki & 1];
#pragma unroll
    for (int cb = 0; cb < 4; ++cb) {
      v16bf bf = frag_b_lds(buf, cb, lane);
      acc[cb] = __builtin_amdgcn_wmma_f32_16x16x32_bf16(
          false, a, false, bf, (short)0, acc[cb], false, false);
    }
    __syncthreads();
  }
  int c = lane & 15, h = lane >> 4;
  bool rm[4];
#pragma unroll
  for (int cb = 0; cb < 4; ++cb) rm[cb] = r_ids[b * NN + n0 + cb * 16 + c] != 0;
#pragma unroll
  for (int v = 0; v < 8; ++v) {
    int m = m0 + v + 8 * h;
    bool cmk = c_ids[b * MM + m] != 0;
#pragma unroll
    for (int cb = 0; cb < 4; ++cb) {
      int n = n0 + cb * 16 + c;
      float x = acc[cb][v] * INV_SCALE;
      att[((size_t)b * MM + m) * NN + n] = (cmk && rm[cb]) ? x : NEGBIG;
    }
  }
}

// ---------------- stage 2: softmax (wave per row / per column) ----------------
__global__ __launch_bounds__(256) void row_softmax_kernel(
    const float* __restrict__ att, u16* __restrict__ Pc) {
  int wid = blockIdx.x * 8 + (threadIdx.x >> 5);
  int lane = threadIdx.x & 31;
  int b = wid >> 9, m = wid & 511;
  const float* row = att + ((size_t)b * MM + m) * NN;
  float x[16], mx = -3.4e38f;
#pragma unroll
  for (int t = 0; t < 16; ++t) { x[t] = row[lane + 32 * t]; mx = fmaxf(mx, x[t]); }
  for (int s = 16; s > 0; s >>= 1) mx = fmaxf(mx, __shfl_xor(mx, s));
  float sum = 0.f;
#pragma unroll
  for (int t = 0; t < 16; ++t) { x[t] = __expf(x[t] - mx); sum += x[t]; }
  for (int s = 16; s > 0; s >>= 1) sum += __shfl_xor(sum, s);
  float inv = 1.0f / sum;
  u16* o = Pc + ((size_t)b * MM + m) * NN;
#pragma unroll
  for (int t = 0; t < 16; ++t) o[lane + 32 * t] = f2bf(x[t] * inv);
}

__global__ __launch_bounds__(256) void col_softmax_kernel(
    const float* __restrict__ att, u16* __restrict__ PrT) {
  int wid = blockIdx.x * 8 + (threadIdx.x >> 5);
  int lane = threadIdx.x & 31;
  int b = wid >> 9, n = wid & 511;
  const float* base = att + (size_t)b * MM * NN + n;
  float x[16], mx = -3.4e38f;
#pragma unroll
  for (int t = 0; t < 16; ++t) { x[t] = base[(size_t)(lane + 32 * t) * NN]; mx = fmaxf(mx, x[t]); }
  for (int s = 16; s > 0; s >>= 1) mx = fmaxf(mx, __shfl_xor(mx, s));
  float sum = 0.f;
#pragma unroll
  for (int t = 0; t < 16; ++t) { x[t] = __expf(x[t] - mx); sum += x[t]; }
  for (int s = 16; s > 0; s >>= 1) sum += __shfl_xor(sum, s);
  float inv = 1.0f / sum;
  u16* o = PrT + ((size_t)b * NN + n) * MM;
#pragma unroll
  for (int t = 0; t < 16; ++t) o[lane + 32 * t] = f2bf(x[t] * inv);
}

// ---------------- stage 3: hat = P . V  (V from K-contiguous transposed copy) --
__global__ __launch_bounds__(256) void pv_kernel(
    const u16* __restrict__ P, const u16* __restrict__ T, u16* __restrict__ out) {
  __shared__ alignas(16) u16 bstage[2][BTILE];
  int b = blockIdx.z;
  int tid = threadIdx.x;
  int lane = tid & 31, w = tid >> 5;
  int m0 = blockIdx.x * 128 + w * 16;
  int e0 = blockIdx.y * 64;
  const u16* A    = P + (size_t)b * 512 * 512 + (size_t)m0 * 512;
  const u16* Bcol = T + (size_t)b * EE * 512 + (size_t)e0 * 512;
  v8f z = {0.f,0.f,0.f,0.f,0.f,0.f,0.f,0.f};
  v8f acc[4] = {z, z, z, z};
  const int KIT = 512 / 32;  // 16
  stage_b(Bcol, 512, 0, bstage[0], tid);
  for (int ki = 0; ki < KIT; ++ki) {
    int k0 = ki * 32;
    if (ki + 1 < KIT) { stage_b(Bcol, 512, k0 + 32, bstage[(ki + 1) & 1], tid); wait_async_le1(); }
    else              { wait_async_0(); }
    __syncthreads();
    v16bf a = frag_a_row(A, 512, k0, lane);
    const u16* buf = bstage[ki & 1];
#pragma unroll
    for (int cb = 0; cb < 4; ++cb) {
      v16bf bf = frag_b_lds(buf, cb, lane);
      acc[cb] = __builtin_amdgcn_wmma_f32_16x16x32_bf16(
          false, a, false, bf, (short)0, acc[cb], false, false);
    }
    __syncthreads();
  }
  int c = lane & 15, h = lane >> 4;
#pragma unroll
  for (int v = 0; v < 8; ++v) {
    int m = m0 + v + 8 * h;
#pragma unroll
    for (int cb = 0; cb < 4; ++cb) {
      int e = e0 + cb * 16 + c;
      out[((size_t)b * 512 + m) * EE + e] = f2bf(acc[cb][v]);
    }
  }
}

// ---------------- stage 4: x = relu(submult(hat,hs) @ W_pre + b) * mask -------
__global__ __launch_bounds__(256) void prernn_kernel(
    const u16* __restrict__ hat, const u16* __restrict__ hs,
    const u16* __restrict__ WT, const float* __restrict__ bias,
    const int* __restrict__ ids, float* __restrict__ out) {
  __shared__ alignas(16) u16 bstage[2][BTILE];
  int b = blockIdx.z;
  int tid = threadIdx.x;
  int lane = tid & 31, w = tid >> 5;
  int m0 = blockIdx.x * 128 + w * 16;
  int e0 = blockIdx.y * 64;
  const u16* hat_row = hat + (size_t)b * 512 * EE + (size_t)m0 * EE;
  const u16* hs_row  = hs  + (size_t)b * 512 * EE + (size_t)m0 * EE;
  const u16* Bcol    = WT + (size_t)e0 * FIVE_E;
  v8f z = {0.f,0.f,0.f,0.f,0.f,0.f,0.f,0.f};
  v8f acc[4] = {z, z, z, z};
  const int KIT = FIVE_E / 32;  // 120
  stage_b(Bcol, FIVE_E, 0, bstage[0], tid);
  for (int ki = 0; ki < KIT; ++ki) {
    int k0 = ki * 32;
    __builtin_prefetch(hat_row + ((size_t)(lane & 15)) * EE + (k0 + 32) % EE, 0, 0);
    if (ki + 1 < KIT) { stage_b(Bcol, FIVE_E, k0 + 32, bstage[(ki + 1) & 1], tid); wait_async_le1(); }
    else              { wait_async_0(); }
    __syncthreads();
    v16bf a = frag_a_submult(hat_row, hs_row, k0, lane);
    const u16* buf = bstage[ki & 1];
#pragma unroll
    for (int cb = 0; cb < 4; ++cb) {
      v16bf bf = frag_b_lds(buf, cb, lane);
      acc[cb] = __builtin_amdgcn_wmma_f32_16x16x32_bf16(
          false, a, false, bf, (short)0, acc[cb], false, false);
    }
    __syncthreads();
  }
  int c = lane & 15, h = lane >> 4;
#pragma unroll
  for (int v = 0; v < 8; ++v) {
    int m = m0 + v + 8 * h;
    float msk = (ids[b * 512 + m] != 0) ? 1.0f : 0.0f;
#pragma unroll
    for (int cb = 0; cb < 4; ++cb) {
      int e = e0 + cb * 16 + c;
      float val = acc[cb][v] + bias[e];
      val = val > 0.f ? val : 0.f;
      out[((size_t)b * 512 + m) * EE + e] = val * msk;
    }
  }
}

// ---------------- stage 5: per-batch features [first, mean, max] -------------
__global__ __launch_bounds__(256) void feat_kernel(
    const float* __restrict__ c_x, const float* __restrict__ r_x,
    const int* __restrict__ c_ids, const int* __restrict__ r_ids,
    float* __restrict__ c_feat, float* __restrict__ r_feat) {
  int side = blockIdx.z, b = blockIdx.y;
  int e = blockIdx.x * 256 + threadIdx.x;
  const float* x  = side ? r_x : c_x;
  const int* ids  = side ? r_ids : c_ids;
  float* f        = side ? r_feat : c_feat;
  const float* xb = x + (size_t)b * 512 * EE;
  float first = xb[e];
  float sum = 0.f, mx = -3.4e38f;
  int cnt = 0;
  for (int i = 0; i < 512; ++i) {
    float v = xb[(size_t)i * EE + e];
    sum += v;
    mx = fmaxf(mx, v);
    cnt += (ids[b * 512 + i] != 0) ? 1 : 0;
  }
  f[(size_t)b * THREE_E + e]          = first;
  f[(size_t)b * THREE_E + EE + e]     = sum / (float)cnt;
  f[(size_t)b * THREE_E + 2 * EE + e] = mx;
}

// ---------------- stage 6: h = relu(submult(cf,rf) @ W_ff1 + b1) -------------
__global__ __launch_bounds__(256) void ff1_kernel(
    const float* __restrict__ c_feat, const float* __restrict__ r_feat,
    const float* __restrict__ W1, const float* __restrict__ b1,
    float* __restrict__ hbuf) {
  int b = blockIdx.y;
  int e = blockIdx.x * 256 + threadIdx.x;
  const float* cf = c_feat + (size_t)b * THREE_E;
  const float* rf = r_feat + (size_t)b * THREE_E;
  float acc = 0.f;
  int s = 0, u = 0;
  for (int t = 0; t < FIFT_E; ++t) {
    float a = cf[u], bb = rf[u];
    float v;
    if      (s == 0) v = a;
    else if (s == 1) v = bb;
    else if (s == 2) v = a - bb;
    else if (s == 3) v = a * bb;
    else { float d = a - bb; v = d * d; }
    acc += v * W1[(size_t)t * EE + e];
    if (++u == THREE_E) { u = 0; ++s; }
  }
  acc += b1[e];
  hbuf[(size_t)b * EE + e] = acc > 0.f ? acc : 0.f;
}

// ---------------- stage 7: scores = h @ W_ff2 + b2 ---------------------------
__global__ __launch_bounds__(256) void score_kernel(
    const float* __restrict__ hbuf, const float* __restrict__ W2,
    const float* __restrict__ b2, float* __restrict__ out) {
  int b = blockIdx.x;
  int lane = threadIdx.x & 31, w = threadIdx.x >> 5;
  float acc = 0.f;
  for (int e = threadIdx.x; e < EE; e += 256) acc += hbuf[(size_t)b * EE + e] * W2[e];
  for (int s = 16; s > 0; s >>= 1) acc += __shfl_xor(acc, s);
  __shared__ float ls[8];
  if (lane == 0) ls[w] = acc;
  __syncthreads();
  if (threadIdx.x == 0) {
    float t = 0.f;
    for (int i = 0; i < 8; ++i) t += ls[i];
    out[b] = t + b2[0];
  }
}

// ---------------- host-side launcher ----------------
extern "C" void kernel_launch(void* const* d_in, const int* in_sizes, int n_in,
                              void* d_out, int out_size, void* d_ws, size_t ws_size,
                              hipStream_t stream) {
  (void)in_sizes; (void)n_in; (void)out_size; (void)ws_size;
  const int*   c_ids = (const int*)d_in[0];
  const int*   r_ids = (const int*)d_in[1];
  const float* c_hs  = (const float*)d_in[2];
  const float* r_hs  = (const float*)d_in[3];
  const float* W_pre = (const float*)d_in[4];
  const float* b_pre = (const float*)d_in[5];
  const float* W_ff1 = (const float*)d_in[6];
  const float* b_ff1 = (const float*)d_in[7];
  const float* W_ff2 = (const float*)d_in[8];
  const float* b_ff2 = (const float*)d_in[9];
  float* out = (float*)d_out;

  // workspace layout (256B aligned slabs)
  char* w = (char*)d_ws;
  size_t off = 0;
  auto alloc = [&](size_t bytes) {
    void* p = w + off;
    off += (bytes + 255) & ~(size_t)255;
    return p;
  };
  const size_t nME = (size_t)BB * MM * EE;   // 12.58M elems
  const size_t nMN = (size_t)BB * MM * NN;   // 8.39M elems
  u16* c_bf  = (u16*)alloc(nME * 2);
  u16* r_bf  = (u16*)alloc(nME * 2);
  u16* cT    = (u16*)alloc(nME * 2);
  u16* rT    = (u16*)alloc(nME * 2);
  u16* WT    = (u16*)alloc((size_t)EE * FIVE_E * 2);
  float* att = (float*)alloc(nMN * 4);
  u16* Pc    = (u16*)alloc(nMN * 2);
  u16* PrT   = (u16*)alloc(nMN * 2);
  u16* c_hat = (u16*)alloc(nME * 2);
  u16* r_hat = (u16*)alloc(nME * 2);
  float* c_x = (float*)alloc(nME * 4);
  float* r_x = (float*)alloc(nME * 4);
  float* c_feat = (float*)alloc((size_t)BB * THREE_E * 4);
  float* r_feat = (float*)alloc((size_t)BB * THREE_E * 4);
  float* hbuf   = (float*)alloc((size_t)BB * EE * 4);

  // 0) precision conversion + transposes
  convert_kernel<<<dim3((unsigned)(nME / 256), 2), 256, 0, stream>>>(
      c_hs, r_hs, c_bf, r_bf, cT, rT);
  wpret_kernel<<<(EE * FIVE_E) / 256, 256, 0, stream>>>(W_pre, WT);

  // 1) attention scores (WMMA + async-LDS B staging)
  att_kernel<<<dim3(MM / 128, NN / 64, BB), 256, 0, stream>>>(
      c_bf, r_bf, c_ids, r_ids, att);

  // 2) softmaxes
  row_softmax_kernel<<<(BB * MM) / 8, 256, 0, stream>>>(att, Pc);
  col_softmax_kernel<<<(BB * NN) / 8, 256, 0, stream>>>(att, PrT);

  // 3) weighted projections (WMMA + async-LDS B staging)
  pv_kernel<<<dim3(MM / 128, EE / 64, BB), 256, 0, stream>>>(Pc, rT, c_hat);
  pv_kernel<<<dim3(NN / 128, EE / 64, BB), 256, 0, stream>>>(PrT, cT, r_hat);

  // 4) submult + pre_rnn (WMMA, dominant cost)
  prernn_kernel<<<dim3(MM / 128, EE / 64, BB), 256, 0, stream>>>(
      c_hat, c_bf, WT, b_pre, c_ids, c_x);
  prernn_kernel<<<dim3(NN / 128, EE / 64, BB), 256, 0, stream>>>(
      r_hat, r_bf, WT, b_pre, r_ids, r_x);

  // 5) per-batch pooled features
  feat_kernel<<<dim3(EE / 256, BB, 2), 256, 0, stream>>>(
      c_x, r_x, c_ids, r_ids, c_feat, r_feat);

  // 6) final MLP
  ff1_kernel<<<dim3(EE / 256, BB), 256, 0, stream>>>(c_feat, r_feat, W_ff1, b_ff1, hbuf);
  score_kernel<<<BB, 256, 0, stream>>>(hbuf, W_ff2, b_ff2, out);
}
